// RNNDecoder_88115549045304
// MI455X (gfx1250) — matrix operand order repaired and checked by reading, not compiled, not run on hardware
//
#include <hip/hip_runtime.h>
#include <cstdint>
#include <cstddef>

// Problem constants (from reference)
#define B_    32
#define T_    256
#define H_    1024
#define E_    1024
#define ENC_  512
#define H4_   4096
#define MROWS (B_ * T_)   // 8192
#define KCOMB (E_ + H_)   // 2048

typedef __attribute__((ext_vector_type(16))) __bf16 v16bf;
typedef __attribute__((ext_vector_type(8)))  float  v8f;

__device__ __forceinline__ v8f wmma_bf16(v16bf a, v16bf b, v8f c) {
  // (neg_a, A, neg_b, B, c_mod, C, reuse_a, reuse_b)
  return __builtin_amdgcn_wmma_f32_16x16x32_bf16(false, a, false, b, (short)0, c,
                                                 false, false);
}

union FragU { uint4 u[2]; v16bf v; };

// A-matrix 16x32 bf16 fragment (ISA 7.12.2): lane l%16 = row M, lanes<16 hold
// K[0..8)+K[16..24), lanes>=16 hold K[8..16)+K[24..32).
__device__ __forceinline__ v16bf load_a_frag(const __bf16* A, int lda, int m0,
                                             int k0, int lane) {
  const __bf16* p = A + (m0 + (lane & 15)) * lda + k0 + ((lane >> 4) << 3);
  FragU t;
  t.u[0] = *reinterpret_cast<const uint4*>(p);
  t.u[1] = *reinterpret_cast<const uint4*>(p + 16);
  return t.v;
}

// B-matrix 32x16 bf16 fragment where B[k][n] = W[n][k], W row-major (N,K):
// lane l%16 = column N (weight row), lanes<16 hold K[0..8)+K[16..24), lanes>=16
// hold K[8..16)+K[24..32) (same k order as the A fragment).
__device__ __forceinline__ v16bf load_b_frag(const __bf16* W, int ldw, int n0,
                                             int k0, int lane) {
  const __bf16* p = W + (n0 + (lane & 15)) * ldw + k0 + ((lane >> 4) << 3);
  FragU t;
  t.u[0] = *reinterpret_cast<const uint4*>(p);
  t.u[1] = *reinterpret_cast<const uint4*>(p + 16);
  return t.v;
}

__device__ __forceinline__ float sigmoidf_(float x) {
  return 1.0f / (1.0f + __expf(-x));
}

// ---------------------------------------------------------------------------
// fp32 -> bf16 conversion (weights / activations)
// ---------------------------------------------------------------------------
__global__ void cvt_f32_bf16_kernel(const float* __restrict__ src,
                                    __bf16* __restrict__ dst, int n) {
  for (int i = blockIdx.x * blockDim.x + threadIdx.x; i < n;
       i += gridDim.x * blockDim.x)
    dst[i] = (__bf16)src[i];
}

__global__ void add_f32_kernel(const float* __restrict__ a,
                               const float* __restrict__ b,
                               float* __restrict__ o, int n) {
  int i = blockIdx.x * blockDim.x + threadIdx.x;
  if (i < n) o[i] = a[i] + b[i];
}

// ---------------------------------------------------------------------------
// Attention (pre-time, 'general'): one block per batch row. Tiny FLOPs.
// ---------------------------------------------------------------------------
__global__ __launch_bounds__(256) void attention_kernel(
    const float* __restrict__ enc_out,   // (B, ENC, H)
    const float* __restrict__ enc_h,     // (B, L, H) batch-first
    const unsigned char* __restrict__ mask,  // (B, ENC)
    const float* __restrict__ attn_W,    // (H, H)
    float* __restrict__ ctx) {           // (B, H)
  const int b = blockIdx.x;
  const int tid = threadIdx.x;
  __shared__ float lh[H_];
  __shared__ float hid[H_];
  __shared__ float w[ENC_];
  __shared__ float red[256];

  const float* hrow = enc_h + (b * 2 + 1) * H_;  // last layer hidden
  for (int i = tid; i < H_; i += 256) lh[i] = hrow[i];
  __syncthreads();

  for (int j = tid; j < H_; j += 256) {
    const float* wr = attn_W + j * H_;
    float s = 0.f;
    for (int k = 0; k < H_; ++k) s += lh[k] * wr[k];
    hid[j] = s;
  }
  __syncthreads();

  const float* eb = enc_out + b * ENC_ * H_;
  for (int s = tid; s < ENC_; s += 256) {
    const float* er = eb + s * H_;
    float sc = 0.f;
    for (int k = 0; k < H_; ++k) sc += hid[k] * er[k];
    w[s] = mask[b * ENC_ + s] ? sc : -1e20f;
  }
  __syncthreads();

  // softmax over ENC_=512 with 256 threads
  red[tid] = fmaxf(w[tid], w[tid + 256]);
  __syncthreads();
  for (int off = 128; off > 0; off >>= 1) {
    if (tid < off) red[tid] = fmaxf(red[tid], red[tid + off]);
    __syncthreads();
  }
  const float mx = red[0];
  __syncthreads();
  float e0 = __expf(w[tid] - mx), e1 = __expf(w[tid + 256] - mx);
  w[tid] = e0;
  w[tid + 256] = e1;
  red[tid] = e0 + e1;
  __syncthreads();
  for (int off = 128; off > 0; off >>= 1) {
    if (tid < off) red[tid] += red[tid + off];
    __syncthreads();
  }
  const float inv = 1.0f / red[0];
  __syncthreads();

  for (int j = tid; j < H_; j += 256) {
    float a = 0.f;
    for (int s = 0; s < ENC_; ++s) a += w[s] * eb[s * H_ + j];
    ctx[b * H_ + j] = a * inv;
  }
}

// ---------------------------------------------------------------------------
// merged[r, k] = k < E ? emb[xs[r], k] : ctx[b, k-E]   (bf16 out)
// ---------------------------------------------------------------------------
__global__ void build_merged_kernel(const int* __restrict__ xs,
                                    const float* __restrict__ emb,
                                    const float* __restrict__ ctx,
                                    __bf16* __restrict__ merged) {
  int idx = blockIdx.x * blockDim.x + threadIdx.x;  // MROWS*KCOMB = 16M
  if (idx >= MROWS * KCOMB) return;
  const int r = idx >> 11;       // row = b*T + t
  const int k = idx & (KCOMB - 1);
  const int b = r >> 8;
  float v;
  if (k < E_) {
    const long tok = (long)xs[r];
    v = emb[tok * E_ + k];
  } else {
    v = ctx[b * H_ + (k - E_)];
  }
  merged[idx] = (__bf16)v;
}

// ---------------------------------------------------------------------------
// Generic WMMA GEMM: C(M,N) = A(M,K) * W(N,K)^T, A/W bf16.
// Block tile 128x128, 8 waves = 4(M) x 2(N); wave tile 32x64 (2x4 wmma tiles).
// Inner K loop is software-pipelined (double-buffered fragments) so the
// next K-slab's global loads overlap the current WMMA chain.
// TANH_BF16_OUT: out = tanh(acc) as bf16 (combine path); else fp32 out.
// ---------------------------------------------------------------------------
template <bool TANH_BF16_OUT>
__global__ __launch_bounds__(256) void gemm_bf16_kernel(
    const __bf16* __restrict__ A, const __bf16* __restrict__ W,
    float* __restrict__ Cf, __bf16* __restrict__ Cb, int M, int N, int K) {
  const int tid = threadIdx.x;
  const int lane = tid & 31;
  const int wave = tid >> 5;
  const int wm = wave & 3;
  const int wn = wave >> 2;
  const int m0 = blockIdx.x * 128 + wm * 32;
  const int n0 = blockIdx.y * 128 + wn * 64;

  v8f acc[2][4] = {};

  // prologue: fragments for k = 0
  v16bf a0 = load_a_frag(A, K, m0, 0, lane);
  v16bf a1 = load_a_frag(A, K, m0 + 16, 0, lane);
  v16bf b0 = load_b_frag(W, K, n0, 0, lane);
  v16bf b1 = load_b_frag(W, K, n0 + 16, 0, lane);
  v16bf b2 = load_b_frag(W, K, n0 + 32, 0, lane);
  v16bf b3 = load_b_frag(W, K, n0 + 48, 0, lane);

  for (int k = 32; k <= K; k += 32) {
    v16bf na0, na1, nb0, nb1, nb2, nb3;
    if (k < K) {
      na0 = load_a_frag(A, K, m0, k, lane);
      na1 = load_a_frag(A, K, m0 + 16, k, lane);
      nb0 = load_b_frag(W, K, n0, k, lane);
      nb1 = load_b_frag(W, K, n0 + 16, k, lane);
      nb2 = load_b_frag(W, K, n0 + 32, k, lane);
      nb3 = load_b_frag(W, K, n0 + 48, k, lane);
    }
    acc[0][0] = wmma_bf16(a0, b0, acc[0][0]);
    acc[1][0] = wmma_bf16(a1, b0, acc[1][0]);
    acc[0][1] = wmma_bf16(a0, b1, acc[0][1]);
    acc[1][1] = wmma_bf16(a1, b1, acc[1][1]);
    acc[0][2] = wmma_bf16(a0, b2, acc[0][2]);
    acc[1][2] = wmma_bf16(a1, b2, acc[1][2]);
    acc[0][3] = wmma_bf16(a0, b3, acc[0][3]);
    acc[1][3] = wmma_bf16(a1, b3, acc[1][3]);
    a0 = na0; a1 = na1;
    b0 = nb0; b1 = nb1; b2 = nb2; b3 = nb3;
  }

  // C layout (ISA 7.12.2): VGPR i -> row = i + 8*(lane/16), col = lane%16
  const int rbase = m0 + ((lane >> 4) << 3);
  const int cbase = n0 + (lane & 15);
#pragma unroll
  for (int mt = 0; mt < 2; ++mt) {
#pragma unroll
    for (int nt = 0; nt < 4; ++nt) {
#pragma unroll
      for (int i = 0; i < 8; ++i) {
        const int r = rbase + mt * 16 + i;
        const int c = cbase + nt * 16;
        if (TANH_BF16_OUT) {
          Cb[r * N + c] = (__bf16)tanhf(acc[mt][nt][i]);
        } else {
          Cf[r * N + c] = acc[mt][nt][i];
        }
      }
    }
  }
}

// ---------------------------------------------------------------------------
// One LSTM timestep: gates = G[:,t,:] + h_{t-1} @ Whh^T + bias, cell update.
// Grid: 16 blocks (each owns output cols j in [nb*64, nb*64+64)).
// 8 waves: wave&1 = M-tile (batch 0..15 / 16..31), wave>>1 = gate (i,f,g,o).
// Recurrent matmul via WMMA bf16 (double-buffered K loop); gate slabs staged
// through LDS; then fused sigmoid/tanh cell update. h ping-pongs between
// bf16 buffers across steps (cross-step dependency = stream ordering).
// ---------------------------------------------------------------------------
__global__ __launch_bounds__(256) void lstm_step_kernel(
    const float* __restrict__ G,        // (B*T, 4H)  input projections
    const __bf16* __restrict__ Whh,     // (4H, H) bf16
    const float* __restrict__ bias,     // (4H) = bih + bhh
    const __bf16* __restrict__ h_in,    // (B, H) bf16, h_{t-1}
    __bf16* __restrict__ h_out,         // (B, H) bf16, h_t
    float* __restrict__ h_state,        // (B, H) fp32, h_t
    float* __restrict__ c_state,        // (B, H) fp32, in/out
    __bf16* __restrict__ out_b,         // (B*T, H) bf16 or null
    float* __restrict__ out_f,          // (B*T, H) fp32 or null
    int t) {
  __shared__ float g_lds[4][B_][68];  // padded to dodge bank conflicts

  const int nb = blockIdx.x;
  const int tid = threadIdx.x;
  const int lane = tid & 31;
  const int wave = tid >> 5;
  const int mt = wave & 1;      // batch tile
  const int gate = wave >> 1;   // 0=i 1=f 2=g 3=o
  const int ncol = gate * H_ + nb * 64;

  v8f acc0 = {}, acc1 = {}, acc2 = {}, acc3 = {};

  v16bf a  = load_a_frag(h_in, H_, mt * 16, 0, lane);
  v16bf b0 = load_b_frag(Whh, H_, ncol, 0, lane);
  v16bf b1 = load_b_frag(Whh, H_, ncol + 16, 0, lane);
  v16bf b2 = load_b_frag(Whh, H_, ncol + 32, 0, lane);
  v16bf b3 = load_b_frag(Whh, H_, ncol + 48, 0, lane);

  for (int k = 32; k <= H_; k += 32) {
    v16bf na, nb0, nb1, nb2, nb3;
    if (k < H_) {
      na  = load_a_frag(h_in, H_, mt * 16, k, lane);
      nb0 = load_b_frag(Whh, H_, ncol, k, lane);
      nb1 = load_b_frag(Whh, H_, ncol + 16, k, lane);
      nb2 = load_b_frag(Whh, H_, ncol + 32, k, lane);
      nb3 = load_b_frag(Whh, H_, ncol + 48, k, lane);
    }
    acc0 = wmma_bf16(a, b0, acc0);
    acc1 = wmma_bf16(a, b1, acc1);
    acc2 = wmma_bf16(a, b2, acc2);
    acc3 = wmma_bf16(a, b3, acc3);
    a = na;
    b0 = nb0; b1 = nb1; b2 = nb2; b3 = nb3;
  }

  const int mloc = mt * 16 + ((lane >> 4) << 3);
  const int jloc = lane & 15;
#pragma unroll
  for (int i = 0; i < 8; ++i) {
    g_lds[gate][mloc + i][jloc]      = acc0[i];
    g_lds[gate][mloc + i][jloc + 16] = acc1[i];
    g_lds[gate][mloc + i][jloc + 32] = acc2[i];
    g_lds[gate][mloc + i][jloc + 48] = acc3[i];
  }
  __syncthreads();

  // fused cell update: 32 batches x 64 cols = 2048 outputs / 256 threads
#pragma unroll
  for (int idx = tid; idx < B_ * 64; idx += 256) {
    const int b = idx >> 6;
    const int jj = idx & 63;
    const int j = nb * 64 + jj;
    const int grow = (b * T_ + t) * H4_;
    const float iv = g_lds[0][b][jj] + G[grow + j]           + bias[j];
    const float fv = g_lds[1][b][jj] + G[grow + H_ + j]      + bias[H_ + j];
    const float gv = g_lds[2][b][jj] + G[grow + 2 * H_ + j]  + bias[2 * H_ + j];
    const float ov = g_lds[3][b][jj] + G[grow + 3 * H_ + j]  + bias[3 * H_ + j];
    const float c = sigmoidf_(fv) * c_state[b * H_ + j] +
                    sigmoidf_(iv) * tanhf(gv);
    const float h = sigmoidf_(ov) * tanhf(c);
    c_state[b * H_ + j] = c;
    h_state[b * H_ + j] = h;
    h_out[b * H_ + j] = (__bf16)h;
    const int orow = (b * T_ + t) * H_ + j;
    if (out_b) out_b[orow] = (__bf16)h;
    if (out_f) out_f[orow] = h;
  }
}

// ---------------------------------------------------------------------------
// Per-layer state init from encoder hidden/cell (batch-first (B, L, H)).
// ---------------------------------------------------------------------------
__global__ void init_layer_kernel(const float* __restrict__ enc_h,
                                  const float* __restrict__ enc_c, int l,
                                  float* __restrict__ hst,
                                  float* __restrict__ cst,
                                  __bf16* __restrict__ hb) {
  int i = blockIdx.x * blockDim.x + threadIdx.x;
  if (i >= B_ * H_) return;
  const int b = i >> 10;
  const int j = i & (H_ - 1);
  const float h = enc_h[(b * 2 + l) * H_ + j];
  const float c = enc_c[(b * 2 + l) * H_ + j];
  hst[i] = h;
  cst[i] = c;
  hb[i] = (__bf16)h;
}

// ---------------------------------------------------------------------------
// Pack final h/c states into d_out tail as batch-first (B, L, H).
// ---------------------------------------------------------------------------
__global__ void finalize_kernel(const float* __restrict__ h0,
                                const float* __restrict__ h1,
                                const float* __restrict__ c0,
                                const float* __restrict__ c1,
                                float* __restrict__ out_tail) {
  int i = blockIdx.x * blockDim.x + threadIdx.x;
  if (i >= B_ * 2 * H_) return;
  const int b = i >> 11;
  const int l = (i >> 10) & 1;
  const int j = i & (H_ - 1);
  const float* h = l ? h1 : h0;
  const float* c = l ? c1 : c0;
  out_tail[i] = h[b * H_ + j];
  out_tail[B_ * 2 * H_ + i] = c[b * H_ + j];
}

// ---------------------------------------------------------------------------
extern "C" void kernel_launch(void* const* d_in, const int* in_sizes, int n_in,
                              void* d_out, int out_size, void* d_ws,
                              size_t ws_size, hipStream_t stream) {
  const int* xs = (const int*)d_in[0];
  const float* enc_out = (const float*)d_in[1];
  const float* enc_h = (const float*)d_in[2];
  const float* enc_c = (const float*)d_in[3];
  const unsigned char* attn_mask = (const unsigned char*)d_in[4];
  const float* emb = (const float*)d_in[5];
  const float* attn_W = (const float*)d_in[6];
  const float* combine_W = (const float*)d_in[7];
  const float* Wih0 = (const float*)d_in[8];
  const float* Whh0 = (const float*)d_in[9];
  const float* bih0 = (const float*)d_in[10];
  const float* bhh0 = (const float*)d_in[11];
  const float* Wih1 = (const float*)d_in[12];
  const float* Whh1 = (const float*)d_in[13];
  const float* bih1 = (const float*)d_in[14];
  const float* bhh1 = (const float*)d_in[15];
  float* out = (float*)d_out;

  // workspace carve-up
  char* ws = (char*)d_ws;
  size_t off = 0;
  auto alloc = [&](size_t bytes) -> char* {
    off = (off + 255) & ~(size_t)255;
    char* p = ws + off;
    off += bytes;
    return p;
  };
  __bf16* wih0_b = (__bf16*)alloc((size_t)H4_ * H_ * 2);
  __bf16* whh0_b = (__bf16*)alloc((size_t)H4_ * H_ * 2);
  __bf16* wih1_b = (__bf16*)alloc((size_t)H4_ * H_ * 2);
  __bf16* whh1_b = (__bf16*)alloc((size_t)H4_ * H_ * 2);
  __bf16* comb_b = (__bf16*)alloc((size_t)E_ * KCOMB * 2);
  float* bias0 = (float*)alloc(H4_ * 4);
  float* bias1 = (float*)alloc(H4_ * 4);
  float* ctx = (float*)alloc((size_t)B_ * H_ * 4);
  __bf16* merged_b = (__bf16*)alloc((size_t)MROWS * KCOMB * 2);
  __bf16* xes_b = (__bf16*)alloc((size_t)MROWS * E_ * 2);
  __bf16* y0_b = (__bf16*)alloc((size_t)MROWS * H_ * 2);
  float* G = (float*)alloc((size_t)MROWS * H4_ * 4);
  float* hst0 = (float*)alloc((size_t)B_ * H_ * 4);
  float* hst1 = (float*)alloc((size_t)B_ * H_ * 4);
  float* cst0 = (float*)alloc((size_t)B_ * H_ * 4);
  float* cst1 = (float*)alloc((size_t)B_ * H_ * 4);
  __bf16* hb0 = (__bf16*)alloc((size_t)B_ * H_ * 2);
  __bf16* hb1 = (__bf16*)alloc((size_t)B_ * H_ * 2);

  // 1) weight conversion to bf16
  cvt_f32_bf16_kernel<<<2048, 256, 0, stream>>>(Wih0, wih0_b, H4_ * H_);
  cvt_f32_bf16_kernel<<<2048, 256, 0, stream>>>(Whh0, whh0_b, H4_ * H_);
  cvt_f32_bf16_kernel<<<2048, 256, 0, stream>>>(Wih1, wih1_b, H4_ * H_);
  cvt_f32_bf16_kernel<<<2048, 256, 0, stream>>>(Whh1, whh1_b, H4_ * H_);
  cvt_f32_bf16_kernel<<<2048, 256, 0, stream>>>(combine_W, comb_b, E_ * KCOMB);
  add_f32_kernel<<<16, 256, 0, stream>>>(bih0, bhh0, bias0, H4_);
  add_f32_kernel<<<16, 256, 0, stream>>>(bih1, bhh1, bias1, H4_);

  // 2) attention (fixed across timesteps with attn_time='pre')
  attention_kernel<<<B_, 256, 0, stream>>>(enc_out, enc_h, attn_mask, attn_W, ctx);

  // 3) merged = [emb[xs], ctx] in bf16
  build_merged_kernel<<<(MROWS * KCOMB) / 256, 256, 0, stream>>>(xs, emb, ctx,
                                                                 merged_b);

  // 4) xes = tanh(merged @ combine_W^T)  (8192 x 2048 x 1024, WMMA, tanh->bf16)
  gemm_bf16_kernel<true><<<dim3(MROWS / 128, E_ / 128), 256, 0, stream>>>(
      merged_b, comb_b, nullptr, xes_b, MROWS, E_, KCOMB);

  // 5) layer 0 input projections: G = xes @ Wih0^T  (8192 x 1024 x 4096)
  gemm_bf16_kernel<false><<<dim3(MROWS / 128, H4_ / 128), 256, 0, stream>>>(
      xes_b, wih0_b, G, nullptr, MROWS, H4_, H_);

  // 6) layer 0 recurrence
  init_layer_kernel<<<(B_ * H_) / 256, 256, 0, stream>>>(enc_h, enc_c, 0, hst0,
                                                         cst0, hb0);
  for (int t = 0; t < T_; ++t) {
    const __bf16* hi = (t & 1) ? hb1 : hb0;
    __bf16* ho = (t & 1) ? hb0 : hb1;
    lstm_step_kernel<<<16, 256, 0, stream>>>(G, whh0_b, bias0, hi, ho, hst0,
                                             cst0, y0_b, nullptr, t);
  }

  // 7) layer 1 input projections: G = y0 @ Wih1^T (reuse G buffer)
  gemm_bf16_kernel<false><<<dim3(MROWS / 128, H4_ / 128), 256, 0, stream>>>(
      y0_b, wih1_b, G, nullptr, MROWS, H4_, H_);

  // 8) layer 1 recurrence, y -> d_out
  init_layer_kernel<<<(B_ * H_) / 256, 256, 0, stream>>>(enc_h, enc_c, 1, hst1,
                                                         cst1, hb0);
  for (int t = 0; t < T_; ++t) {
    const __bf16* hi = (t & 1) ? hb1 : hb0;
    __bf16* ho = (t & 1) ? hb0 : hb1;
    lstm_step_kernel<<<16, 256, 0, stream>>>(G, whh1_b, bias1, hi, ho, hst1,
                                             cst1, nullptr, out, t);
  }

  // 9) final h/c states (batch-first (B, L, H)) into d_out tail
  finalize_kernel<<<(B_ * 2 * H_) / 256, 256, 0, stream>>>(
      hst0, hst1, cst0, cst1, out + (size_t)MROWS * H_);
}
